// CustomModel_91259465106013
// MI455X (gfx1250) — matrix operand order repaired and checked by reading, not compiled, not run on hardware
//
#include <hip/hip_runtime.h>

// ---------------------------------------------------------------------------
// VGG16 gradient-energy softmax pipeline for gfx1250 (MI455X).
// Convs = implicit GEMM on v_wmma_f32_16x16x32_bf16 (wave32).
//  - Activations NHWC bf16: B fragment = two aligned uint4 loads.
//  - Weights repacked K' = (ky*3+kx)*IC + ic (layer0 rows zero-padded to 32).
//  - A fragments global->registers; B tile double-buffered in LDS; global
//    loads and LDS reads software-pipelined around 16 WMMA per K-step.
// ---------------------------------------------------------------------------

typedef __attribute__((ext_vector_type(16))) __bf16 v16bf;
typedef __attribute__((ext_vector_type(8)))  float  v8f;

#define BATCH 8

__device__ __forceinline__ unsigned short f2bf(float f) {
  unsigned int u = __float_as_uint(f);
  unsigned int r = u + 0x7FFFu + ((u >> 16) & 1u);   // round-to-nearest-even
  return (unsigned short)(r >> 16);
}
__device__ __forceinline__ float bf2f(unsigned short h) {
  return __uint_as_float(((unsigned int)h) << 16);
}

union BfFrag {
  v16bf v;
  unsigned short u[16];
  uint4 q[2];          // two 16-byte chunks
};

union HalfPack {
  uint4 q;
  unsigned short u[8];
};

// ---------------- packing / conversion kernels ----------------

// OIHW f32 -> [OC][Kpad] bf16 with K' = (ky*3+kx)*IC + ic, zero pad to Kpad.
__global__ void k_pack_w(const float* __restrict__ in,
                         unsigned short* __restrict__ out,
                         int OC, int IC, int Kpad) {
  int Ktot = IC * 9;
  int total = OC * Kpad;
  int i = blockIdx.x * blockDim.x + threadIdx.x;
  if (i >= total) return;
  int oc = i / Kpad;
  int kk = i - oc * Kpad;
  unsigned short v = 0;
  if (kk < Ktot) {
    int kykx = kk / IC;
    int ic   = kk - kykx * IC;
    v = f2bf(in[(oc * IC + ic) * 9 + kykx]);
  }
  out[i] = v;
}

// S [B,3,H,W] f32 -> act [B,H,W,3] bf16
__global__ void k_to_nhwc(const float* __restrict__ in,
                          unsigned short* __restrict__ out, int HW) {
  int total = BATCH * HW * 3;
  int i = blockIdx.x * blockDim.x + threadIdx.x;
  if (i >= total) return;
  int c = i % 3;
  int pix = i / 3;
  int b = pix / HW;
  int p = pix - b * HW;
  out[i] = f2bf(in[(b * 3 + c) * HW + p]);
}

__global__ void k_zero(float* __restrict__ p, int n) {
  int i = blockIdx.x * blockDim.x + threadIdx.x;
  if (i < n) p[i] = 0.0f;
}

// ---------------- fast implicit-GEMM conv (IC % 32 == 0) ----------------
// actIn : [B,H,W,IC] bf16 ; Wp: packed [OC, 9*IC] bf16 ; actOut: [B,H,W,OC]
// Tile: (64*MPW) OC x 128 pixels, 128 threads (4 waves).

template <int MPW>
__global__ void __launch_bounds__(128)
k_conv3x3_wmma_fast(const unsigned short* __restrict__ actIn,
                    const unsigned short* __restrict__ Wp,
                    const float* __restrict__ bias,
                    unsigned short* __restrict__ actOut,
                    const unsigned short* __restrict__ zeroPad,
                    int IC, int OC, int H, int W) {
  const int Ktot = IC * 9;
  const int HW   = H * W;
  const int Npix = BATCH * HW;
  const int tid  = threadIdx.x;
  const int tileN = blockIdx.x * 128;
  const int tileM = blockIdx.y * (64 * MPW);

  __shared__ uint4 lBq[2][16][32];        // [buf][ns*2+chunk][lane]

  const int wv   = tid >> 5;
  const int lane = tid & 31;
  const int half = lane >> 4;
  const int l16  = lane & 15;

  // A rows this wave owns (fragment layout == two contiguous 16B row spans)
  const unsigned short* a_row[MPW];
#pragma unroll
  for (int mp = 0; mp < MPW; ++mp)
    a_row[mp] = Wp + (size_t)(tileM + (wv + 4 * mp) * 16 + l16) * Ktot + half * 16;

  // pixel decode hoisted out of the K loop (fragments ns = wv, wv+4)
  int  pixOff[2];
  int  py[2], px[2];
  bool pok[2];
#pragma unroll
  for (int s = 0; s < 2; ++s) {
    int ng = tileN + (wv + 4 * s) * 16 + l16;
    bool v = ng < Npix;
    int ngc = v ? ng : 0;
    int b = ngc / HW;
    int p = ngc - b * HW;
    int y = p / W, x = p - y * W;
    pixOff[s] = ngc;
    py[s] = y; px[s] = x; pok[s] = v;
  }

  BfFrag breg[2];   // gathered B fragments (registers)
  auto gatherB = [&](int icb_, int dy_, int dx_, int dOff_) {
#pragma unroll
    for (int s = 0; s < 2; ++s) {
      const int ic0 = icb_ + 16 * half;
      int iy = py[s] + dy_, ix = px[s] + dx_;
      bool ok = pok[s] & ((unsigned)iy < (unsigned)H) & ((unsigned)ix < (unsigned)W);
      const uint4* src = ok
          ? (const uint4*)(actIn + ((long long)(pixOff[s] + dOff_) * IC + ic0))
          : (const uint4*)zeroPad;
      breg[s].q[0] = src[0];
      breg[s].q[1] = src[1];
    }
  };
  auto storeB = [&](int bufi) {
#pragma unroll
    for (int s = 0; s < 2; ++s) {
      const int ns = wv + 4 * s;
      lBq[bufi][ns * 2 + 0][lane] = breg[s].q[0];
      lBq[bufi][ns * 2 + 1][lane] = breg[s].q[1];
    }
  };

  v8f acc[MPW][8] = {};

  int icb = 0, kykx = 0, dy = -1, dx = -1, dOff = -W - 1;

  // ---- prologue: stage k0 = 0 into buf 0, load first A fragments ----
  gatherB(icb, dy, dx, dOff);
  storeB(0);
  BfFrag Acur[MPW], Anext[MPW];
#pragma unroll
  for (int mp = 0; mp < MPW; ++mp) {
    Acur[mp].q[0] = *(const uint4*)(a_row[mp]);
    Acur[mp].q[1] = *(const uint4*)(a_row[mp] + 8);
  }
  __syncthreads();

  int buf = 0;
  for (int k0 = 0; k0 < Ktot; k0 += 32) {
    const bool more = (k0 + 32) < Ktot;
    // advance tap state to k0+32 (scalar/uniform)
    icb += 32;
    if (icb == IC) {
      icb = 0;
      ++kykx;
      dy = kykx / 3 - 1;
      dx = kykx - (kykx / 3) * 3 - 1;
      dOff = dy * W + dx;
    }
    // issue next-step global loads; latency hidden behind WMMAs below
    if (more) {
#pragma unroll
      for (int mp = 0; mp < MPW; ++mp) {
        Anext[mp].q[0] = *(const uint4*)(a_row[mp] + (k0 + 32));
        Anext[mp].q[1] = *(const uint4*)(a_row[mp] + (k0 + 32) + 8);
      }
      gatherB(icb, dy, dx, dOff);
    }
    // compute: ping-pong B fragment regs so LDS reads lead the WMMA by one.
    BfFrag Bf[2];
    Bf[0].q[0] = lBq[buf][0][lane];
    Bf[0].q[1] = lBq[buf][1][lane];
#pragma unroll
    for (int ns = 0; ns < 8; ++ns) {
      const int cn = ns & 1;
      if (ns < 7) {
        Bf[cn ^ 1].q[0] = lBq[buf][(ns + 1) * 2 + 0][lane];
        Bf[cn ^ 1].q[1] = lBq[buf][(ns + 1) * 2 + 1][lane];
      }
#pragma unroll
      for (int mp = 0; mp < MPW; ++mp)
        acc[mp][ns] = __builtin_amdgcn_wmma_f32_16x16x32_bf16(
            false, Acur[mp].v, false, Bf[cn].v, (short)0, acc[mp][ns], false, false);
    }
    // commit next B tile, rotate A registers
    if (more) {
      storeB(buf ^ 1);
#pragma unroll
      for (int mp = 0; mp < MPW; ++mp) Acur[mp] = Anext[mp];
    }
    __syncthreads();
    buf ^= 1;
  }

  // ---- epilogue: bias + ReLU, vectorized NHWC store (8 oc = one uint4) ----
#pragma unroll
  for (int mp = 0; mp < MPW; ++mp) {
    const int ocb = tileM + (wv + 4 * mp) * 16 + half * 8;
    const float4 b0 = *(const float4*)(bias + ocb);
    const float4 b1 = *(const float4*)(bias + ocb + 4);
    const float bv[8] = {b0.x, b0.y, b0.z, b0.w, b1.x, b1.y, b1.z, b1.w};
#pragma unroll
    for (int ns = 0; ns < 8; ++ns) {
      int ng = tileN + ns * 16 + l16;
      if (ng >= Npix) continue;
      HalfPack pk;
#pragma unroll
      for (int r = 0; r < 8; ++r) {
        float val = acc[mp][ns][r] + bv[r];
        val = val > 0.0f ? val : 0.0f;
        pk.u[r] = f2bf(val);
      }
      *(uint4*)(actOut + ((long long)ng * OC + ocb)) = pk.q;
    }
  }
}

// ---------------- layer conv1_1 (IC=3): branch-free via K padded to 32 ------
// Wp rows have stride 32, zero-filled beyond K=27 -> unconditional A loads and
// no k-guard on B (padded A zeros annihilate don't-care B values).
__global__ void __launch_bounds__(128)
k_conv3x3_wmma_gen(const unsigned short* __restrict__ actIn,
                   const unsigned short* __restrict__ Wp,
                   const float* __restrict__ bias,
                   unsigned short* __restrict__ actOut,
                   const unsigned short* __restrict__ zeroPad,
                   int IC, int OC, int H, int W) {
  const int Ktot = IC * 9;          // 27
  const int Kpad = 32;
  const int HW   = H * W;
  const int Npix = BATCH * HW;
  const int tid  = threadIdx.x;
  const int tileN = blockIdx.x * 128;
  const int tileM = blockIdx.y * 64;

  __shared__ uint4 lBq[16][32];

  const int wv   = tid >> 5;
  const int lane = tid & 31;
  const int half = lane >> 4;
  const int l16  = lane & 15;

  const unsigned short* a_row =
      Wp + (size_t)(tileM + wv * 16 + l16) * Kpad + half * 16;

  v8f acc[8] = {};

  for (int k0 = 0; k0 < Ktot; k0 += 32) {
    // A fragment: unconditional (rows zero-padded to Kpad)
    BfFrag A;
    A.q[0] = *(const uint4*)(a_row + k0);
    A.q[1] = *(const uint4*)(a_row + k0 + 8);

#pragma unroll
    for (int s = 0; s < 2; ++s) {
      const int ns = wv + 4 * s;
      const int ng = tileN + ns * 16 + l16;
      const int kbase = k0 + 16 * half;
      BfFrag tmp;
      bool epn = ng < Npix;
      int ngc = epn ? ng : 0;
      int b = ngc / HW;
      int p = ngc - b * HW;
      int y = p / W, x = p - y * W;
      const long long pixBase = (long long)ngc;
#pragma unroll
      for (int e = 0; e < 16; ++e) {
        int k = kbase + e;
        int kc = k > 26 ? 26 : k;          // padded-K values are don't-care
        int kykx = kc / IC;
        int ic = kc - kykx * IC;
        int dy = kykx / 3 - 1;
        int dx = kykx - (kykx / 3) * 3 - 1;
        int iy = y + dy, ix = x + dx;
        bool ok = epn & ((unsigned)iy < (unsigned)H) & ((unsigned)ix < (unsigned)W);
        const unsigned short* src =
            ok ? (actIn + (pixBase + dy * W + dx) * IC + ic) : zeroPad;
        tmp.u[e] = *src;
      }
      lBq[ns * 2 + 0][lane] = tmp.q[0];
      lBq[ns * 2 + 1][lane] = tmp.q[1];
    }
    __syncthreads();

#pragma unroll
    for (int ns = 0; ns < 8; ++ns) {
      BfFrag Bf;
      Bf.q[0] = lBq[ns * 2 + 0][lane];
      Bf.q[1] = lBq[ns * 2 + 1][lane];
      acc[ns] = __builtin_amdgcn_wmma_f32_16x16x32_bf16(
          false, A.v, false, Bf.v, (short)0, acc[ns], false, false);
    }
    __syncthreads();
  }

  const int ocb = tileM + wv * 16 + half * 8;
  const float4 b0 = *(const float4*)(bias + ocb);
  const float4 b1 = *(const float4*)(bias + ocb + 4);
  const float bv[8] = {b0.x, b0.y, b0.z, b0.w, b1.x, b1.y, b1.z, b1.w};
#pragma unroll
  for (int ns = 0; ns < 8; ++ns) {
    int ng = tileN + ns * 16 + l16;
    if (ng >= Npix) continue;
    HalfPack pk;
#pragma unroll
    for (int r = 0; r < 8; ++r) {
      float val = acc[ns][r] + bv[r];
      val = val > 0.0f ? val : 0.0f;
      pk.u[r] = f2bf(val);
    }
    *(uint4*)(actOut + ((long long)ng * OC + ocb)) = pk.q;
  }
}

// ---------------- 2x2 maxpool, stride 2 (NHWC) ----------------
__global__ void k_maxpool2(const unsigned short* __restrict__ in,
                           unsigned short* __restrict__ out,
                           int C, int H, int W) {
  int Ho = H >> 1, Wo = W >> 1;
  int total = BATCH * Ho * Wo * C;
  int i = blockIdx.x * blockDim.x + threadIdx.x;
  if (i >= total) return;
  int c = i % C; int t = i / C;
  int xo = t % Wo; t /= Wo;
  int yo = t % Ho; int b = t / Ho;
  const unsigned short* pl =
      in + ((long long)((b * H + yo * 2) * W + xo * 2)) * C + c;
  int sW = W * C;
  float a = bf2f(pl[0]), b2 = bf2f(pl[C]);
  float c2 = bf2f(pl[sW]), d = bf2f(pl[sW + C]);
  out[i] = f2bf(fmaxf(fmaxf(a, b2), fmaxf(c2, d)));
}

// ---------------- depthwise Laplacian grad-energy sum (NHWC) ----------------
__global__ void __launch_bounds__(256)
k_grad_energy(const unsigned short* __restrict__ act,
              float* __restrict__ accum, int C, int H, int W) {
  int i = blockIdx.x * 256 + threadIdx.x;
  int pix = i / C;
  int x = pix % W; int t = pix / W;
  int y = t % H;  int b = t / H;
  float s = 0.0f;
#pragma unroll
  for (int dy = -1; dy <= 1; ++dy)
#pragma unroll
    for (int dx = -1; dx <= 1; ++dx) {
      int yy = y + dy, xx = x + dx;
      if (yy >= 0 && yy < H && xx >= 0 && xx < W) {
        float vv = bf2f(act[i + (dy * W + dx) * C]);
        s += (dy == 0 && dx == 0) ? -1.0f * vv : 0.125f * vv;
      }
    }
  float sq = s * s;
  __shared__ float red[256];
  red[threadIdx.x] = sq;
  __syncthreads();
#pragma unroll
  for (int s2 = 128; s2 > 0; s2 >>= 1) {
    if (threadIdx.x < s2) red[threadIdx.x] += red[threadIdx.x + s2];
    __syncthreads();
  }
  if (threadIdx.x == 0) atomicAdd(&accum[b], red[0]);
}

// ---------------- finalize ----------------
__global__ void k_finalize(const float* __restrict__ energy,
                           float* __restrict__ out) {
  if (threadIdx.x != 0 || blockIdx.x != 0) return;
  const float inv[5] = {1.0f / (64.0f * 224 * 224), 1.0f / (128.0f * 112 * 112),
                        1.0f / (256.0f * 56 * 56),  1.0f / (512.0f * 28 * 28),
                        1.0f / (512.0f * 14 * 14)};
  float s[16];
  for (int i = 0; i < 16; ++i) {
    int inp = i >> 3, b = i & 7;
    float m = 0.0f;
    for (int f = 0; f < 5; ++f) m += energy[inp * 40 + f * 8 + b] * inv[f];
    s[i] = m * 0.2f;
  }
  float mx = s[0];
  for (int i = 1; i < 16; ++i) mx = fmaxf(mx, s[i]);
  float sum = 0.0f;
  for (int i = 0; i < 16; ++i) { s[i] = expf(s[i] - mx); sum += s[i]; }
  for (int i = 0; i < 16; ++i) out[i] = s[i] / sum;
}

// ---------------------------------------------------------------------------
extern "C" void kernel_launch(void* const* d_in, const int* in_sizes, int n_in,
                              void* d_out, int out_size, void* d_ws, size_t ws_size,
                              hipStream_t stream) {
  (void)in_sizes; (void)n_in; (void)out_size; (void)ws_size;

  static const int IC[13] = {3, 64, 64, 128, 128, 256, 256, 256, 512, 512, 512, 512, 512};
  static const int OC[13] = {64, 64, 128, 128, 256, 256, 256, 512, 512, 512, 512, 512, 512};
  static const int HH[13] = {224, 224, 112, 112, 56, 56, 56, 28, 28, 28, 14, 14, 14};
  static const bool blockEnd[13] = {false, true, false, true, false, false, true,
                                    false, false, true, false, false, true};

  char* ws = (char*)d_ws;
  size_t off = 0;
  auto carve = [&](size_t bytes) -> char* {
    char* p = ws + off;
    off = (off + bytes + 255) & ~(size_t)255;
    return p;
  };

  // layer 0 gets K padded to 32; others use Ktot (already multiple of 32)
  int Kpad[13];
  for (int l = 0; l < 13; ++l) Kpad[l] = (l == 0) ? 32 : IC[l] * 9;

  unsigned short* wbf[13];
  for (int l = 0; l < 13; ++l)
    wbf[l] = (unsigned short*)carve((size_t)OC[l] * Kpad[l] * 2);
  const size_t maxAct = (size_t)64 * BATCH * 224 * 224;      // elements
  unsigned short* actA = (unsigned short*)carve(maxAct * 2);
  unsigned short* actB = (unsigned short*)carve(maxAct * 2);
  float* misc = (float*)carve(256 * sizeof(float));
  float* energy = misc;                                 // 80 floats used
  unsigned short* zeroPad = (unsigned short*)(misc + 128);  // 512 zero bytes
  // total workspace: ~132 MB

  for (int l = 0; l < 13; ++l) {
    int n = OC[l] * Kpad[l];
    k_pack_w<<<(n + 255) / 256, 256, 0, stream>>>(
        (const float*)d_in[2 + 2 * l], wbf[l], OC[l], IC[l], Kpad[l]);
  }
  k_zero<<<1, 256, 0, stream>>>(misc, 256);

  for (int inp = 0; inp < 2; ++inp) {
    const float* S = (const float*)d_in[inp];
    {
      int n = BATCH * 224 * 224 * 3;
      k_to_nhwc<<<(n + 255) / 256, 256, 0, stream>>>(S, actA, 224 * 224);
    }
    unsigned short* cur = actA;
    unsigned short* oth = actB;
    int feat = 0;
    for (int l = 0; l < 13; ++l) {
      int H = HH[l], W = HH[l];
      int Npix = BATCH * H * W;
      const float* bias = (const float*)d_in[3 + 2 * l];
      if (l == 0) {
        dim3 grid((Npix + 127) / 128, OC[l] / 64);
        k_conv3x3_wmma_gen<<<grid, 128, 0, stream>>>(
            cur, wbf[l], bias, oth, zeroPad, IC[l], OC[l], H, W);
      } else if ((OC[l] & 127) == 0) {
        dim3 grid((Npix + 127) / 128, OC[l] / 128);
        k_conv3x3_wmma_fast<2><<<grid, 128, 0, stream>>>(
            cur, wbf[l], bias, oth, zeroPad, IC[l], OC[l], H, W);
      } else {
        dim3 grid((Npix + 127) / 128, OC[l] / 64);
        k_conv3x3_wmma_fast<1><<<grid, 128, 0, stream>>>(
            cur, wbf[l], bias, oth, zeroPad, IC[l], OC[l], H, W);
      }
      { unsigned short* t = cur; cur = oth; oth = t; }

      if (blockEnd[l]) {
        int n = BATCH * H * W * OC[l];               // multiple of 256
        k_grad_energy<<<n / 256, 256, 0, stream>>>(
            cur, energy + inp * 40 + feat * 8, OC[l], H, W);
        ++feat;
        if (l != 12) {
          int np = BATCH * (H / 2) * (W / 2) * OC[l];
          k_maxpool2<<<(np + 255) / 256, 256, 0, stream>>>(cur, oth, OC[l], H, W);
          { unsigned short* t = cur; cur = oth; oth = t; }
        }
      }
    }
  }
  k_finalize<<<1, 32, 0, stream>>>(energy, (float*)d_out);
}
